// CausalSelfAttention_6073083757241
// MI455X (gfx1250) — compile-verified, matrix operand order checked by reading, not run on hardware
//
#include <hip/hip_runtime.h>

// ---------------------------------------------------------------------------
// Types
// ---------------------------------------------------------------------------
typedef __attribute__((ext_vector_type(16))) __bf16 v16bf;
typedef __attribute__((ext_vector_type(8)))  float  v8f;
typedef __attribute__((ext_vector_type(4))) unsigned int u32x4;
typedef __attribute__((ext_vector_type(2))) unsigned int u32x2;
typedef __attribute__((ext_vector_type(4)))  float  f32x4;
typedef int i32x4v __attribute__((vector_size(16)));  // matches builtin's V4i

union FragU { v16bf v; u32x4 u[2]; };

// Problem constants (B=2, T=2048, C=1024, H=16, D=64)
#define CB 2
#define CT 2048
#define CC 1024
#define CH 16
#define CD 64

// ---------------------------------------------------------------------------
// Async global->LDS copy (CDNA5 GLOBAL_LOAD_ASYNC_TO_LDS_B128, ASYNCcnt).
// Guarded: falls back to a synchronous b128 copy if the builtin is absent.
// ---------------------------------------------------------------------------
#if defined(__gfx1250__) && __has_builtin(__builtin_amdgcn_global_load_async_to_lds_b128)
#define HAVE_ASYNC_LDS 1
#else
#define HAVE_ASYNC_LDS 0
#endif

__device__ inline void cp_b128(__bf16* lds, const __bf16* g) {
#if HAVE_ASYNC_LDS
  __builtin_amdgcn_global_load_async_to_lds_b128(
      (i32x4v*)(void*)g, (i32x4v*)(void*)lds, 0, 0);
#else
  *(u32x4*)lds = *(const u32x4*)g;
#endif
}

template <int N> __device__ inline void cp_wait() {
#if HAVE_ASYNC_LDS
#if __has_builtin(__builtin_amdgcn_s_wait_asynccnt)
  __builtin_amdgcn_s_wait_asynccnt(N);
#else
  asm volatile("s_wait_asynccnt %0" ::"n"(N));
#endif
#endif
}

// ---------------------------------------------------------------------------
// WMMA helpers. 16-bit A/B fragment (16x32): lane holds row = lane%16,
// K-halves per the CDNA5 ISA layout: VGPR0-3 = K[(lane/16)*8 .. +7],
// VGPR4-7 = K[16 + (lane/16)*8 .. +7]. Two b128 loads per lane.
// ---------------------------------------------------------------------------
__device__ inline v16bf load_frag_b16(const __bf16* base, int stride) {
  const int lane = threadIdx.x & 31;
  const int row  = lane & 15;
  const int koff = (lane >> 4) << 3;
  const __bf16* p = base + row * stride + koff;
  FragU f;
  f.u[0] = *(const u32x4*)(p);
  f.u[1] = *(const u32x4*)(p + 16);
  return f.v;
}

__device__ inline v8f wmma_bf16f32(v16bf a, v16bf b, v8f c) {
  return __builtin_amdgcn_wmma_f32_16x16x32_bf16(
      /*neg_a=*/false, a, /*neg_b=*/false, b,
      /*c_mod=*/(short)0, c, /*reuse_a=*/false, /*reuse_b=*/false);
}

// ---------------------------------------------------------------------------
// fp32 -> bf16 elementwise convert (vectorized x4)
// ---------------------------------------------------------------------------
__global__ void cvt_f32_to_bf16(const float* __restrict__ in,
                                __bf16* __restrict__ out, int n4) {
  int i = blockIdx.x * blockDim.x + threadIdx.x;
  if (i < n4) {
    f32x4 f = ((const f32x4*)in)[i];
    __bf16 t[4] = {(__bf16)f.x, (__bf16)f.y, (__bf16)f.z, (__bf16)f.w};
    *(u32x2*)(out + (size_t)i * 4) = *(u32x2*)t;
  }
}

// ---------------------------------------------------------------------------
// W[K][N] fp32 -> Wt[N][K] bf16 (LDS-tiled transpose)
// ---------------------------------------------------------------------------
__global__ void transpose_cvt_kernel(const float* __restrict__ W,
                                     __bf16* __restrict__ Wt, int K, int N) {
  __shared__ float tile[32][33];
  const int k0 = blockIdx.y * 32, n0 = blockIdx.x * 32;
  const int tx = threadIdx.x, ty = threadIdx.y;
  for (int i = ty; i < 32; i += 8)
    tile[i][tx] = W[(size_t)(k0 + i) * N + n0 + tx];
  __syncthreads();
  for (int i = ty; i < 32; i += 8)
    Wt[(size_t)(n0 + i) * K + k0 + tx] = (__bf16)tile[tx][i];
}

// ---------------------------------------------------------------------------
// Generic bf16 GEMM:  C[m][n] = sum_k A[m][k]*Bt[n][k] + bias[n]
// Block tile 128x128, 8 waves (4M x 2N), wave tile 32x64, k-stage 64,
// double-buffered LDS filled by async global->LDS, 16 WMMAs per stage.
// mode 0: fp32 out.  mode 1: qkv epilogue -> Q(scaled)/K [B,H,T,D], V^T [B,H,D,T]
// ---------------------------------------------------------------------------
#define GBM 128
#define GBN 128
#define GBK 64
#define LDAB 72   // padded stride: (9*r mod 16) permutation -> conflict-free

__global__ __launch_bounds__(256) void gemm_bf16_kernel(
    const __bf16* __restrict__ A, const __bf16* __restrict__ Bt,
    const float* __restrict__ bias, int M, int N, int K, int mode,
    float* __restrict__ outF, __bf16* __restrict__ Qb,
    __bf16* __restrict__ Kbuf, __bf16* __restrict__ Vt) {
  __shared__ alignas(16) __bf16 As[2][GBM * LDAB];
  __shared__ alignas(16) __bf16 Bs[2][GBN * LDAB];
  const int tid  = threadIdx.x;
  const int lane = tid & 31;
  const int wave = tid >> 5;
  const int wm   = wave & 3;   // 0..3  (M dir)
  const int wn   = wave >> 2;  // 0..1  (N dir)
  const int m0   = blockIdx.y * GBM;
  const int n0   = blockIdx.x * GBN;

  // 128 rows x 64 bf16 = 1024 b128 chunks per tile; 4 per thread per matrix.
  auto stage = [&](int buf, int kk) {
#pragma unroll
    for (int i = 0; i < 4; i++) {
      const int idx = tid + i * 256;
      const int r = idx >> 3, c = idx & 7;
      cp_b128(&As[buf][r * LDAB + c * 8],
              A + (size_t)(m0 + r) * K + kk + c * 8);
      cp_b128(&Bs[buf][r * LDAB + c * 8],
              Bt + (size_t)(n0 + r) * K + kk + c * 8);
    }
  };

  v8f acc[2][4];
#pragma unroll
  for (int mi = 0; mi < 2; mi++)
#pragma unroll
    for (int ni = 0; ni < 4; ni++)
      acc[mi][ni] = (v8f){0.f, 0.f, 0.f, 0.f, 0.f, 0.f, 0.f, 0.f};

  stage(0, 0);
  const int ntiles = K / GBK;
  for (int t = 0; t < ntiles; ++t) {
    const int buf = t & 1;
    if (t + 1 < ntiles) {
      stage(buf ^ 1, (t + 1) * GBK);
      cp_wait<8>();   // oldest 8 (stage t) done; stage t+1 stays in flight
    } else {
      cp_wait<0>();
    }
    __syncthreads();
#pragma unroll
    for (int ko = 0; ko < GBK; ko += 32) {
      v16bf af[2], bfrag[4];
#pragma unroll
      for (int mi = 0; mi < 2; mi++)
        af[mi] = load_frag_b16(&As[buf][(wm * 32 + mi * 16) * LDAB + ko], LDAB);
#pragma unroll
      for (int ni = 0; ni < 4; ni++)
        bfrag[ni] = load_frag_b16(&Bs[buf][(wn * 64 + ni * 16) * LDAB + ko], LDAB);
#pragma unroll
      for (int mi = 0; mi < 2; mi++)
#pragma unroll
        for (int ni = 0; ni < 4; ni++)
          acc[mi][ni] = wmma_bf16f32(af[mi], bfrag[ni], acc[mi][ni]);
    }
    __syncthreads();
  }

  // Epilogue. C-fragment layout: VGPR j, lane l -> row j + 8*(l/16), col l%16.
  const int col0 = lane & 15;
  const int rb   = (lane >> 4) * 8;
#pragma unroll
  for (int mi = 0; mi < 2; mi++)
#pragma unroll
    for (int ni = 0; ni < 4; ni++) {
      const int n = n0 + wn * 64 + ni * 16 + col0;
      const float bv = bias[n];
#pragma unroll
      for (int j = 0; j < 8; j++) {
        const int m = m0 + wm * 32 + mi * 16 + rb + j;
        const float val = acc[mi][ni][j] + bv;
        if (mode == 0) {
          outF[(size_t)m * N + n] = val;
        } else {
          const int b = m >> 11, t = m & (CT - 1);
          if (n < CC) {  // Q, pre-scaled by 1/sqrt(D)
            const int h = n >> 6, d = n & 63;
            Qb[(((size_t)b * CH + h) * CT + t) * CD + d] = (__bf16)(val * 0.125f);
          } else if (n < 2 * CC) {  // K
            const int nn = n - CC, h = nn >> 6, d = nn & 63;
            Kbuf[(((size_t)b * CH + h) * CT + t) * CD + d] = (__bf16)val;
          } else {  // V transposed -> [B,H,D,T]
            const int nn = n - 2 * CC, h = nn >> 6, d = nn & 63;
            Vt[(((size_t)b * CH + h) * CD + d) * CT + t] = (__bf16)val;
          }
        }
      }
    }
}

// ---------------------------------------------------------------------------
// Flash attention. Grid: (T/64, B*H). 128 threads = 4 waves; wave w owns
// 16 query rows. Key tiles of 32, double-buffered async K/V staging,
// online softmax in C-fragment layout, P transposed C->A layout through
// per-wave LDS scratch (same-wave LDS ops are in-order per ISA).
// ---------------------------------------------------------------------------
#define LDK 72
#define LDV 40
#define LDP 40

__global__ __launch_bounds__(128) void attn_kernel(
    const __bf16* __restrict__ Qb, const __bf16* __restrict__ Kb,
    const __bf16* __restrict__ Vt, __bf16* __restrict__ Yb) {
  __shared__ alignas(16) __bf16 Ks[2][32 * LDK];      // [key][d]
  __shared__ alignas(16) __bf16 Vs[2][CD * LDV];      // [d][key]
  __shared__ alignas(16) __bf16 Ps[4 * 16 * LDP];     // per-wave P scratch
  const int tid = threadIdx.x, lane = tid & 31, wave = tid >> 5;
  const int bh = blockIdx.y;
  const int b = bh >> 4, h = bh & 15;
  const int qb0 = blockIdx.x * 64;
  const int q0 = qb0 + wave * 16;
  const size_t baseQK = (size_t)bh * CT * CD;
  const size_t baseV  = (size_t)bh * CD * CT;

  // K tile 32x64 (256 b128) + V^T tile 64x32 (256 b128): 4 copies/thread.
  auto stageKV = [&](int buf, int k0) {
#pragma unroll
    for (int i = 0; i < 2; i++) {
      const int idx = tid + i * 128;
      const int kr = idx >> 3, kc = idx & 7;
      cp_b128(&Ks[buf][kr * LDK + kc * 8],
              Kb + baseQK + (size_t)(k0 + kr) * CD + kc * 8);
      const int vr = idx >> 2, vc = idx & 3;
      cp_b128(&Vs[buf][vr * LDV + vc * 8],
              Vt + baseV + (size_t)vr * CT + k0 + vc * 8);
    }
  };

  // Loop-invariant Q fragments (Q pre-scaled by 1/sqrt(D))
  const v16bf qa0 = load_frag_b16(Qb + baseQK + (size_t)q0 * CD, CD);
  const v16bf qa1 = load_frag_b16(Qb + baseQK + (size_t)q0 * CD + 32, CD);

  v8f o[4];
#pragma unroll
  for (int f = 0; f < 4; f++)
    o[f] = (v8f){0.f, 0.f, 0.f, 0.f, 0.f, 0.f, 0.f, 0.f};
  float rm[8], rs[8];
#pragma unroll
  for (int j = 0; j < 8; j++) { rm[j] = -1e30f; rs[j] = 0.f; }

  const int col0 = lane & 15;
  const int rb   = (lane >> 4) * 8;
  __bf16* Pw = Ps + wave * 16 * LDP;
  const int nkt = (qb0 + 64) >> 5;  // causal bound for the whole block

  stageKV(0, 0);
  for (int kt = 0; kt < nkt; ++kt) {
    const int buf = kt & 1;
    const int k0 = kt * 32;
    if (kt + 1 < nkt) {
      stageKV(buf ^ 1, k0 + 32);
      cp_wait<4>();
    } else {
      cp_wait<0>();
    }
    __syncthreads();

    // S = Q K^T : two 16-key column fragments, reduce over d = 2 k-steps
    v8f s0 = (v8f){0.f, 0.f, 0.f, 0.f, 0.f, 0.f, 0.f, 0.f};
    v8f s1 = (v8f){0.f, 0.f, 0.f, 0.f, 0.f, 0.f, 0.f, 0.f};
    s0 = wmma_bf16f32(qa0, load_frag_b16(&Ks[buf][0], LDK), s0);
    s0 = wmma_bf16f32(qa1, load_frag_b16(&Ks[buf][32], LDK), s0);
    s1 = wmma_bf16f32(qa0, load_frag_b16(&Ks[buf][16 * LDK], LDK), s1);
    s1 = wmma_bf16f32(qa1, load_frag_b16(&Ks[buf][16 * LDK + 32], LDK), s1);

    // Causal mask + online softmax (stats replicated across the 16-lane half)
    float p0[8], p1[8];
#pragma unroll
    for (int j = 0; j < 8; j++) {
      const int qg = q0 + rb + j;
      const float v0 = (k0 + col0      <= qg) ? s0[j] : -1e30f;
      const float v1 = (k0 + col0 + 16 <= qg) ? s1[j] : -1e30f;
      float tm = fmaxf(v0, v1);
#pragma unroll
      for (int off = 1; off < 16; off <<= 1)
        tm = fmaxf(tm, __shfl_xor(tm, off, 32));
      const float nm = fmaxf(rm[j], tm);
      const float alpha = __expf(rm[j] - nm);
      const float e0 = __expf(v0 - nm);
      const float e1 = __expf(v1 - nm);
      float ps = e0 + e1;
#pragma unroll
      for (int off = 1; off < 16; off <<= 1) ps += __shfl_xor(ps, off, 32);
      rs[j] = rs[j] * alpha + ps;
      rm[j] = nm;
#pragma unroll
      for (int f = 0; f < 4; f++) o[f][j] *= alpha;
      p0[j] = e0; p1[j] = e1;
    }

    // C-layout -> A-layout for P via per-wave LDS scratch
#pragma unroll
    for (int j = 0; j < 8; j++) {
      Pw[(rb + j) * LDP + col0]      = (__bf16)p0[j];
      Pw[(rb + j) * LDP + col0 + 16] = (__bf16)p1[j];
    }
    asm volatile("" ::: "memory");  // keep LDS write->read program order
    const v16bf pa = load_frag_b16(Pw, LDP);

    // O += P V : 4 d-column fragments
#pragma unroll
    for (int f = 0; f < 4; f++)
      o[f] = wmma_bf16f32(pa, load_frag_b16(&Vs[buf][f * 16 * LDV], LDV), o[f]);
    __syncthreads();
  }

  // Normalize and write y as bf16 in [B, T, H*D] for the projection GEMM
#pragma unroll
  for (int f = 0; f < 4; f++)
#pragma unroll
    for (int j = 0; j < 8; j++) {
      const int t = q0 + rb + j;
      const int d = f * 16 + col0;
      Yb[((size_t)b * CT + t) * (CH * CD) + h * CD + d] =
          (__bf16)(o[f][j] / rs[j]);
    }
}

// ---------------------------------------------------------------------------
// Launch
// ---------------------------------------------------------------------------
extern "C" void kernel_launch(void* const* d_in, const int* in_sizes, int n_in,
                              void* d_out, int out_size, void* d_ws,
                              size_t ws_size, hipStream_t stream) {
  const float* x      = (const float*)d_in[0];
  const float* W_attn = (const float*)d_in[1];
  const float* b_attn = (const float*)d_in[2];
  const float* W_proj = (const float*)d_in[3];
  const float* b_proj = (const float*)d_in[4];
  float* out = (float*)d_out;

  const int M = CB * CT;  // 4096
  char* ws = (char*)d_ws;
  size_t off = 0;
  auto alloc = [&](size_t bytes) -> void* {
    void* p = ws + off;
    off += (bytes + 255) & ~(size_t)255;
    return p;
  };
  __bf16* Xb  = (__bf16*)alloc((size_t)M * CC * 2);        // x in bf16
  __bf16* WtA = (__bf16*)alloc((size_t)3 * CC * CC * 2);   // W_attn^T bf16
  __bf16* WtP = (__bf16*)alloc((size_t)CC * CC * 2);       // W_proj^T bf16
  __bf16* Qb  = (__bf16*)alloc((size_t)M * CC * 2);        // [B,H,T,D]
  __bf16* Kb  = (__bf16*)alloc((size_t)M * CC * 2);        // [B,H,T,D]
  __bf16* Vt  = (__bf16*)alloc((size_t)M * CC * 2);        // [B,H,D,T]
  __bf16* Yb  = (__bf16*)alloc((size_t)M * CC * 2);        // [B,T,C]

  // 1) x -> bf16
  {
    const int n4 = M * CC / 4;
    cvt_f32_to_bf16<<<(n4 + 255) / 256, 256, 0, stream>>>(x, Xb, n4);
  }
  // 2) weights -> transposed bf16
  transpose_cvt_kernel<<<dim3(3 * CC / 32, CC / 32), dim3(32, 8), 0, stream>>>(
      W_attn, WtA, CC, 3 * CC);
  transpose_cvt_kernel<<<dim3(CC / 32, CC / 32), dim3(32, 8), 0, stream>>>(
      W_proj, WtP, CC, CC);
  // 3) qkv GEMM with fused split/scale/transpose epilogue
  gemm_bf16_kernel<<<dim3(3 * CC / GBN, M / GBM), 256, 0, stream>>>(
      Xb, WtA, b_attn, M, 3 * CC, CC, /*mode=*/1, nullptr, Qb, Kb, Vt);
  // 4) causal flash attention
  attn_kernel<<<dim3(CT / 64, CB * CH), 128, 0, stream>>>(Qb, Kb, Vt, Yb);
  // 5) output projection
  gemm_bf16_kernel<<<dim3(CC / GBN, M / GBM), 256, 0, stream>>>(
      Yb, WtP, b_proj, M, CC, CC, /*mode=*/0, out, nullptr, nullptr, nullptr);
}
